// VisionUnit_26130581029096
// MI455X (gfx1250) — compile-verified, matrix operand order checked by reading, not run on hardware
//
#include <hip/hip_runtime.h>
#include <hip/hip_bf16.h>

// ---------------------------------------------------------------------------
// VisionUnit cross-attention for gfx1250 (MI455X), bf16 WMMA path.
//
// Sizes: D_IN=768, H=512, E=64, B=32, P=197 (padded to 208 = 13*16), RHO=4.
// Math simplification: mean_p(context) = sum_q w[q]*v[q] with
//   w[q] = (1/197) * sum_p softmax_row_p(q)  -> no [E,B,P,P] materialization.
//
// Data movement plan (23.3 TB/s HBM, 192 MB L2 -> all tensors L2-resident):
//  - proj GEMMs: A fetched as 4x b128 loads/lane; W tile LDS-staged per
//    K-chunk and shared by the 8 waves of a block (one N-tile per block).
//  - attention: 13-wave blocks (one wave per 16-row p-strip); B-operand
//    tiles for the current K-chunk staged to LDS once and consumed by all
//    13 waves (13x L2-traffic reduction vs per-wave global fetch).
// ---------------------------------------------------------------------------

typedef __attribute__((ext_vector_type(16))) __bf16 v16bf;
typedef __attribute__((ext_vector_type(8)))  float  v8f;

#define DIN   768
#define HDIM  512
#define NE    64
#define NB    32
#define NP    197
#define NPT   13      // ceil(197/16) tiles along patch dim
#define NKC   16      // 512/32 K-chunks over H
#define NKCP  24      // 768/32 K-chunks over D_IN
#define RHO_C 4.0f
#define SCALE_C 0.04419417382415922f  // 1/sqrt(512)

// ISA 7.12.2: 16-bit A/B operand element -> K index.
// lane = half*16 + m(or n); element i in v16bf holds K = (i&8)*2 + half*8 + (i&7)
__device__ __forceinline__ int kidx(int i, int half) {
  return ((i & 8) << 1) + (half << 3) + (i & 7);
}

// ---------------------------------------------------------------------------
// Projection GEMM: C[M,512] = X[M,768] * W[768,512] + bias, via bf16 WMMA.
// Block = 8 waves, one N-tile per block, 8 consecutive M-tiles (one per wave).
// W tile for the current K-chunk staged to LDS (operand-lane order, double
// buffered) and shared by all waves.
// mode 0: Q  -> swizzled A-operand tiles  qA[e][pb][kc][lane][16]  (bf16)
// mode 1: K  -> swizzled B-operand tiles  kB[b][nt][kc][lane][16]  (bf16)
// mode 2/3: V / cls_fc -> plain fp32 row-major [M,512]
// ---------------------------------------------------------------------------
__global__ void __launch_bounds__(256)
proj_wmma_kernel(const float* __restrict__ X,
                 const float* __restrict__ W,
                 const float* __restrict__ bias,
                 void* __restrict__ outp,
                 int Mrows, int mode) {
  const int tid  = threadIdx.x;
  const int wave = tid >> 5;
  const int lane = tid & 31;
  const int half = lane >> 4;
  const int ln   = lane & 15;

  const int numMT = Mrows >> 4;          // Mrows is a multiple of 16
  const int tn    = blockIdx.x & 31;     // 512/16 = 32 N-tiles
  const int tm    = (blockIdx.x >> 5) * 8 + wave;
  const bool active = (tm < numMT);      // wave-uniform

  __shared__ __bf16 wTile[2][32][16];    // B-operand tile, double buffered

  // cooperative stage of W[kc*32 .. +31][tn*16 .. +15] into operand order
  auto stageW = [&](int kc, int buf) {
    const int id = tid * 2;              // 256 threads x 2 = 512 elements
    const int K  = id >> 4;
    const int n  = id & 15;              // even
    const float2 wv =
        *(const float2*)&W[(size_t)(kc * 32 + K) * HDIM + tn * 16 + n];
    const int hb = (K >> 3) & 1;
    const int i2 = (K & 7) + ((K >> 4) << 3);
    wTile[buf][hb * 16 + n][i2]     = (__bf16)wv.x;
    wTile[buf][hb * 16 + n + 1][i2] = (__bf16)wv.y;
  };

  v8f acc = {0.f, 0.f, 0.f, 0.f, 0.f, 0.f, 0.f, 0.f};
  const float* Xrow = X + (size_t)(active ? tm * 16 + ln : 0) * DIN;

  stageW(0, 0);
  for (int kc = 0; kc < NKCP; ++kc) {
    __syncthreads();                        // staged buf[kc&1] visible
    if (kc + 1 < NKCP) stageW(kc + 1, (kc + 1) & 1);

    // A operand: K elements i=0..7 and 8..15 are contiguous -> 4x float4
    const float4 f0 = *(const float4*)(Xrow + kc * 32 + half * 8);
    const float4 f1 = *(const float4*)(Xrow + kc * 32 + half * 8 + 4);
    const float4 f2 = *(const float4*)(Xrow + kc * 32 + 16 + half * 8);
    const float4 f3 = *(const float4*)(Xrow + kc * 32 + 16 + half * 8 + 4);
    v16bf a;
    a[0]  = (__bf16)f0.x; a[1]  = (__bf16)f0.y; a[2]  = (__bf16)f0.z; a[3]  = (__bf16)f0.w;
    a[4]  = (__bf16)f1.x; a[5]  = (__bf16)f1.y; a[6]  = (__bf16)f1.z; a[7]  = (__bf16)f1.w;
    a[8]  = (__bf16)f2.x; a[9]  = (__bf16)f2.y; a[10] = (__bf16)f2.z; a[11] = (__bf16)f2.w;
    a[12] = (__bf16)f3.x; a[13] = (__bf16)f3.y; a[14] = (__bf16)f3.z; a[15] = (__bf16)f3.w;

    const v16bf bm = *(const v16bf*)&wTile[kc & 1][lane][0];  // ds_load_b128 x2
    if (active)
      acc = __builtin_amdgcn_wmma_f32_16x16x32_bf16(
          false, a, false, bm, (short)0, acc, false, false);
  }
  if (!active) return;

#pragma unroll
  for (int r = 0; r < 8; ++r) {
    const int row = tm * 16 + r + 8 * half;   // token row
    const int col = tn * 16 + ln;             // h
    const float val = acc[r] + bias[col];
    if (mode >= 2) {
      ((float*)outp)[(size_t)row * HDIM + col] = val;
    } else {
      // scatter into WMMA operand-tile layout (identical formula for A and B;
      // idx16 means m for A-tiles (mode 0), n for B-tiles (mode 1))
      const int seg   = row / NP;             // e / b
      const int p     = row % NP;
      const int bt    = p >> 4;               // pb / nt
      const int idx16 = p & 15;               // m / n
      const int kcc   = col >> 5;
      const int Kw    = col & 31;
      const int hb    = (Kw >> 3) & 1;
      const int i2    = (Kw & 7) + ((Kw >> 4) << 3);
      const int lane2 = hb * 16 + idx16;
      ((__bf16*)outp)[((((size_t)seg * NPT + bt) * NKC + kcc) * 32 + lane2) * 16 + i2] =
          (__bf16)val;
    }
  }
}

// ---------------------------------------------------------------------------
// Block reduction helper for 13-wave32 blocks.
// ---------------------------------------------------------------------------
__device__ __forceinline__ float blockReduceSum13(float v, float* scratch) {
#pragma unroll
  for (int m = 16; m >= 1; m >>= 1) v += __shfl_xor(v, m, 32);
  if ((threadIdx.x & 31) == 0) scratch[threadIdx.x >> 5] = v;
  __syncthreads();
  float t = 0.f;
#pragma unroll
  for (int i = 0; i < NPT; ++i) t += scratch[i];
  __syncthreads();
  return t;
}

// ---------------------------------------------------------------------------
// Attention kernel: one workgroup per (e,b). 416 threads = 13 wave32,
// one wave per 16-row p-strip. Per K-chunk, the 13 B-operand tiles (13 KB)
// are cooperatively staged to LDS (double buffered) and shared by all waves.
// ---------------------------------------------------------------------------
__global__ void __launch_bounds__(416)
attn_kernel(const __bf16* __restrict__ qA,
            const __bf16* __restrict__ kB,
            const float* __restrict__ vf,
            const float* __restrict__ clsfc,
            const float* __restrict__ maskBP,
            const float* __restrict__ ecls,
            const float* __restrict__ mcls,
            const float* __restrict__ lng,
            const float* __restrict__ lnb,
            float* __restrict__ out) {
  const int e = blockIdx.x >> 5;
  const int b = blockIdx.x & 31;
  const int tid  = threadIdx.x;
  const int pb   = tid >> 5;             // wave id == p-strip id, 0..12
  const int lane = tid & 31;
  const int half = lane >> 4;
  const int ln   = lane & 15;

  __shared__ __bf16 kTile[2][NPT][32][16];  // 26.6 KB, double buffered
  __shared__ float wsum[NPT * 16];          // 208 column-sum accumulators
  __shared__ float red[NPT];

  if (tid < NPT * 16) wsum[tid] = 0.f;

  const __bf16* kSrc = kB + (size_t)b * NPT * NKC * 512;  // 512 = 32 lanes * 16
  // cooperative stage: thread t copies 32 B; 13 nt * 32 lanes = 416 threads
  auto stageK = [&](int kc, int buf) {
    const int nt = tid >> 5;
    const int lp = tid & 31;
    const uint4* s = (const uint4*)(kSrc + ((size_t)nt * NKC + kc) * 512 + lp * 16);
    uint4* d = (uint4*)&kTile[buf][nt][lp][0];
    d[0] = s[0];
    d[1] = s[1];
  };

  const v16bf* Abase = (const v16bf*)(qA + ((size_t)e * NPT + pb) * NKC * 512);
  const float* maskp = maskBP + (size_t)b * NP;

  v8f acc[NPT];
  const v8f vzero = {0.f, 0.f, 0.f, 0.f, 0.f, 0.f, 0.f, 0.f};
#pragma unroll
  for (int nt = 0; nt < NPT; ++nt) acc[nt] = vzero;

  // ---- score strip: 16 x 208, accumulated over 16 K-chunks of 32 ----------
  stageK(0, 0);
  for (int kc = 0; kc < NKC; ++kc) {
    __syncthreads();                        // buf[kc&1] staged; prev buf free
    if (kc + 1 < NKC) stageK(kc + 1, (kc + 1) & 1);

    const v16bf a = Abase[kc * 32 + lane];  // contiguous 1 KB/wave
    __builtin_prefetch(Abase + (kc + 1) * 32 + lane, 0, 1);  // global_prefetch_b8
#pragma unroll
    for (int nt = 0; nt < NPT; ++nt) {
      const v16bf bm = *(const v16bf*)&kTile[kc & 1][nt][lane][0];
      acc[nt] = __builtin_amdgcn_wmma_f32_16x16x32_bf16(
          false, a, false, bm, (short)0, acc[nt], false, false);
    }
  }

  // ---- scale + CLS-row rho mask + pad masking (in place) -------------------
  float rmax[8];
#pragma unroll
  for (int r = 0; r < 8; ++r) rmax[r] = -3.0e38f;
#pragma unroll
  for (int nt = 0; nt < NPT; ++nt) {
    const int n = nt * 16 + ln;             // key-patch index q
#pragma unroll
    for (int r = 0; r < 8; ++r) {
      const int prow = pb * 16 + r + 8 * half;
      float s = acc[nt][r] * SCALE_C;
      if (pb == 0 && half == 0 && r == 0 && n < NP)
        s += RHO_C * maskp[n];              // additive mask on CLS query row
      if (n >= NP || prow >= NP) s = -3.0e38f;
      acc[nt][r] = s;
      rmax[r] = fmaxf(rmax[r], s);
    }
  }
  // row max across the 16 lanes of this half (row m lives in one half)
#pragma unroll
  for (int r = 0; r < 8; ++r) {
#pragma unroll
    for (int m = 1; m < 16; m <<= 1)
      rmax[r] = fmaxf(rmax[r], __shfl_xor(rmax[r], m, 16));
  }

  float rsum[8];
#pragma unroll
  for (int r = 0; r < 8; ++r) rsum[r] = 0.f;
#pragma unroll
  for (int nt = 0; nt < NPT; ++nt) {
#pragma unroll
    for (int r = 0; r < 8; ++r) {
      const float ev = __expf(acc[nt][r] - rmax[r]);
      acc[nt][r] = ev;
      rsum[r] += ev;
    }
  }
#pragma unroll
  for (int r = 0; r < 8; ++r) {
#pragma unroll
    for (int m = 1; m < 16; m <<= 1) rsum[r] += __shfl_xor(rsum[r], m, 16);
  }

  float inv[8];
#pragma unroll
  for (int r = 0; r < 8; ++r) inv[r] = 1.0f / rsum[r];

#pragma unroll
  for (int nt = 0; nt < NPT; ++nt) {
    float cs = 0.f;
#pragma unroll
    for (int r = 0; r < 8; ++r) {
      const int prow = pb * 16 + r + 8 * half;
      cs += (prow < NP) ? acc[nt][r] * inv[r] : 0.f;
    }
    cs += __shfl_xor(cs, 16, 32);           // combine both 8-row halves
    if (half == 0) atomicAdd(&wsum[nt * 16 + ln], cs);   // ds_add_f32
  }
  __syncthreads();

  if (tid < NPT * 16) wsum[tid] *= (1.0f / (float)NP);
  __syncthreads();

  // ---- stage 2: ov[h] = sum_q w[q]*v[b,q,h] ; LN ; fused g2l+g2g dot -------
  float a0 = 0.f, a1 = 0.f;
  const int h0 = tid * 2;
  if (tid < 256) {
    const float* vrow = vf + (size_t)b * NP * HDIM + h0;
    for (int q = 0; q < NP; ++q) {
      const float wq = wsum[q];
      const float2 vv = *(const float2*)(vrow + (size_t)q * HDIM);
      a0 += wq * vv.x;
      a1 += wq * vv.y;
    }
  }

  const float tot  = blockReduceSum13(a0 + a1, red);
  const float tot2 = blockReduceSum13(a0 * a0 + a1 * a1, red);
  const float mean = tot * (1.0f / (float)HDIM);
  const float var  = tot2 * (1.0f / (float)HDIM) - mean * mean;
  const float rstd = rsqrtf(var + 1e-5f);

  float pg = 0.f;
  if (tid < 256) {
    const float n0 = (a0 - mean) * rstd * lng[h0]     + lnb[h0];
    const float n1 = (a1 - mean) * rstd * lng[h0 + 1] + lnb[h0 + 1];
    const float* cf = clsfc + (size_t)e * HDIM;
    pg = n0 * cf[h0] + n1 * cf[h0 + 1];
  }

  float gg = 0.f;
  if (tid < 384) {
    const float* mc = mcls + (size_t)b * DIN;
    const float* ec = ecls + (size_t)e * DIN;
    gg = mc[tid] * ec[tid] + mc[tid + 384] * ec[tid + 384];
  }

  const float total = blockReduceSum13(pg + gg, red);
  if (tid == 0) out[(size_t)b * NE + e] = 0.5f * total;
}

// ---------------------------------------------------------------------------
// Launcher
// ---------------------------------------------------------------------------
extern "C" void kernel_launch(void* const* d_in, const int* in_sizes, int n_in,
                              void* d_out, int out_size, void* d_ws, size_t ws_size,
                              hipStream_t stream) {
  const float* ecls = (const float*)d_in[0];   // [64,768]
  const float* etok = (const float*)d_in[1];   // [64,197,768]
  const float* mcls = (const float*)d_in[2];   // [32,768]
  const float* mtok = (const float*)d_in[3];   // [32,197,768]
  const float* mask = (const float*)d_in[4];   // [32,197]
  const float* Wq   = (const float*)d_in[5];
  const float* bq   = (const float*)d_in[6];
  const float* Wk   = (const float*)d_in[7];
  const float* bk   = (const float*)d_in[8];
  const float* Wv   = (const float*)d_in[9];
  const float* bv   = (const float*)d_in[10];
  const float* Wcls = (const float*)d_in[11];
  const float* bcls = (const float*)d_in[12];
  const float* lng  = (const float*)d_in[13];
  const float* lnb  = (const float*)d_in[14];

  // workspace carve-up (~33.5 MB total)
  char* ws = (char*)d_ws;
  const size_t qA_bytes = (size_t)NE * NPT * NKC * 32 * 16 * sizeof(__bf16); // 13.6 MB
  const size_t kB_bytes = (size_t)NB * NPT * NKC * 32 * 16 * sizeof(__bf16); //  6.8 MB
  const size_t vf_bytes = (size_t)NB * NP * HDIM * sizeof(float);            // 12.9 MB
  __bf16* qA   = (__bf16*)(ws);
  __bf16* kBuf = (__bf16*)(ws + qA_bytes);
  float*  vf   = (float*)(ws + qA_bytes + kB_bytes);
  float*  cfc  = (float*)(ws + qA_bytes + kB_bytes + vf_bytes);

  // projections: block = 8 waves (one N-tile, 8 M-tiles), grid = ceil(MT/8)*32
  {
    const int mtQ = (NE * NP) / 16;   // 788
    proj_wmma_kernel<<<((mtQ + 7) / 8) * 32, 256, 0, stream>>>(etok, Wq, bq, qA,
                                                               NE * NP, 0);
    const int mtK = (NB * NP) / 16;   // 394
    proj_wmma_kernel<<<((mtK + 7) / 8) * 32, 256, 0, stream>>>(mtok, Wk, bk, kBuf,
                                                               NB * NP, 1);
    proj_wmma_kernel<<<((mtK + 7) / 8) * 32, 256, 0, stream>>>(mtok, Wv, bv, vf,
                                                               NB * NP, 2);
    const int mtC = NE / 16;          // 4
    proj_wmma_kernel<<<((mtC + 7) / 8) * 32, 256, 0, stream>>>(ecls, Wcls, bcls, cfc,
                                                               NE, 3);
  }

  // fused attention + pooling + LN + similarity: 13 waves per (e,b)
  attn_kernel<<<NE * NB, 416, 0, stream>>>(qA, kBuf, vf, cfc, mask,
                                           ecls, mcls, lng, lnb, (float*)d_out);
}
